// Model_30013231465053
// MI455X (gfx1250) — compile-verified, matrix operand order checked by reading, not run on hardware
//
#include <hip/hip_runtime.h>
#include <hip/hip_bf16.h>
#include <math.h>

typedef __attribute__((ext_vector_type(16))) _Float16 v16h;
typedef __attribute__((ext_vector_type(8)))  _Float16 v8h;
typedef __attribute__((ext_vector_type(8)))  float    v8f;

#define NPTS 2048
#define BATCH 8
#define NGT 16384
#define NOUT 1024

// ---------------------------------------------------------------------------
// f32 -> f16 elementwise (weights converted once per call)
// ---------------------------------------------------------------------------
__global__ __launch_bounds__(256)
void cvt_f16_kernel(const float* __restrict__ in, _Float16* __restrict__ out, int n) {
  const int i = blockIdx.x * 256 + threadIdx.x;
  if (i < n) out[i] = (_Float16)in[i];
}

// ---------------------------------------------------------------------------
// (b,c,n) f32 -> (b,n,c) f16
// ---------------------------------------------------------------------------
__global__ __launch_bounds__(256)
void transpose_cn_kernel(const float* __restrict__ in, _Float16* __restrict__ out,
                         int B, int C, int N) {
  const size_t idx = (size_t)blockIdx.x * 256 + threadIdx.x;
  if (idx >= (size_t)B * C * N) return;
  const int n = (int)(idx % N);
  const size_t t = idx / N;
  const int c = (int)(t % C);
  const int b = (int)(t / C);
  out[((size_t)b * N + n) * C + c] = (_Float16)in[idx];
}

// ---------------------------------------------------------------------------
// Batched WMMA GEMM, all-f16 operands, f32 accumulate:
//   C[b] (f16, MxN) = act(A[b] (f16, MxK) @ W^T (f16, NxK) + bias [+ Res])
// One wave owns a 16x64 C strip: one A fragment feeds 4 independent WMMAs
// (hides XDL latency, 4x A reuse). All operand loads are packed v8h
// (global_load_b128), no converts in the inner loop.
// A-fragment lane layout (16-bit A 16x32): lane m = lane&15,
//   halves h -> k = 16*(h>>3) + 8*(lane>>4) + (h&7); B mirrors with n=lane&15.
// C/D layout: vgpr i -> row m0 + i + 8*(lane>>4), col n0 + (lane&15).
// ---------------------------------------------------------------------------
template <int ACT, bool HAS_RES>
__global__ __launch_bounds__(256)
void wmma_gemm_kernel(const _Float16* __restrict__ A, const _Float16* __restrict__ W,
                      const float* __restrict__ bias, const _Float16* __restrict__ Res,
                      _Float16* __restrict__ C, int M, int N, int K) {
  const int batch = blockIdx.y;
  A += (size_t)batch * M * K;
  C += (size_t)batch * M * N;
  const _Float16* R = HAS_RES ? (Res + (size_t)batch * M * N) : nullptr;

  const int wid  = threadIdx.x >> 5;
  const int lane = threadIdx.x & 31;
  const int hi   = lane >> 4;
  const int lm   = lane & 15;

  const int ntn  = N >> 6;                    // 64-wide N strips
  const int tile = blockIdx.x * 8 + wid;
  const int m0   = (tile / ntn) << 4;
  const int n0   = (tile % ntn) << 6;

  const _Float16* Arow = A + (size_t)(m0 + lm) * K + (hi << 3);
  const _Float16* Wrow = W + (size_t)(n0 + lm) * K + (hi << 3);

  const v8f z = {};
  v8f acc[4] = {z, z, z, z};

  for (int k0 = 0; k0 < K; k0 += 32) {
    if (k0 + 64 < K) __builtin_prefetch(Arow + k0 + 64, 0, 3);
    v16h a;
    {
      const v8h a0 = *(const v8h*)(Arow + k0);
      const v8h a1 = *(const v8h*)(Arow + k0 + 16);
#pragma unroll
      for (int h = 0; h < 8; ++h) { a[h] = a0[h]; a[h + 8] = a1[h]; }
    }
#pragma unroll
    for (int nn = 0; nn < 4; ++nn) {
      const _Float16* wp = Wrow + ((size_t)(nn << 4)) * K + k0;
      const v8h b0 = *(const v8h*)(wp);
      const v8h b1 = *(const v8h*)(wp + 16);
      v16h bm;
#pragma unroll
      for (int h = 0; h < 8; ++h) { bm[h] = b0[h]; bm[h + 8] = b1[h]; }
      acc[nn] = __builtin_amdgcn_wmma_f32_16x16x32_f16(false, a, false, bm,
                                                       (short)0, acc[nn],
                                                       false, false);
    }
  }

#pragma unroll
  for (int nn = 0; nn < 4; ++nn) {
#pragma unroll
    for (int i = 0; i < 8; ++i) {
      const int m = m0 + i + (hi << 3);
      const int n = n0 + (nn << 4) + lm;
      float v = acc[nn][i] + bias[n];
      if (HAS_RES) v += (float)R[(size_t)m * N + n];
      if (ACT == 1) v = 0.5f * v * (1.0f + erff(v * 0.70710678118654752f));
      C[(size_t)m * N + n] = (_Float16)v;
    }
  }
}

// ---------------------------------------------------------------------------
// LayerNorm over last dim C, f16 buffer, f32 math, in place.
// ---------------------------------------------------------------------------
__global__ __launch_bounds__(256)
void layernorm_kernel(_Float16* __restrict__ buf, const float* __restrict__ g,
                      const float* __restrict__ b, int C) {
  __shared__ float red[256];
  const int t = threadIdx.x;
  _Float16* p = buf + (size_t)blockIdx.x * C;

  float s = 0.f;
  for (int i = t; i < C; i += 256) s += (float)p[i];
  red[t] = s; __syncthreads();
  for (int o = 128; o > 0; o >>= 1) { if (t < o) red[t] += red[t + o]; __syncthreads(); }
  const float mu = red[0] / (float)C; __syncthreads();

  float v = 0.f;
  for (int i = t; i < C; i += 256) { const float d = (float)p[i] - mu; v += d * d; }
  red[t] = v; __syncthreads();
  for (int o = 128; o > 0; o >>= 1) { if (t < o) red[t] += red[t + o]; __syncthreads(); }
  const float rstd = rsqrtf(red[0] / (float)C + 1e-5f); __syncthreads();

  for (int i = t; i < C; i += 256)
    p[i] = (_Float16)(((float)p[i] - mu) * rstd * g[i] + b[i]);
}

// ---------------------------------------------------------------------------
// Flash attention (self), NHEAD=4, templated head dim. All-f16 operands.
// qkv: (B, N, 3*C) f16 rows = [q | k | v]; out: (B, N, C) f16.
// Block = 8 waves, each wave one 16-row Q tile of one (b, head).
// S = Q K^T via WMMA chain; online softmax; P relayout D->A through LDS
// (b16 stores in, aligned b128 reads out, explicit s_wait_dscnt); O += P V via WMMA.
// ---------------------------------------------------------------------------
template <int DH>
__global__ __launch_bounds__(256)
void attention_kernel(const _Float16* __restrict__ qkv, _Float16* __restrict__ out,
                      int Npts, float scale) {
  constexpr int C  = 4 * DH;
  constexpr int KC = DH / 32;
  constexpr int DC = DH / 16;
  __shared__ __align__(16) _Float16 pshare[8][16 * 32];

  const int wid  = threadIdx.x >> 5;
  const int lane = threadIdx.x & 31;
  const int hi   = lane >> 4;
  const int lm   = lane & 15;
  const int head = blockIdx.y;
  const int b    = blockIdx.z;
  const int m0   = (blockIdx.x * 8 + wid) << 4;

  const _Float16* base = qkv + (size_t)b * Npts * (3 * C);
  _Float16* pb = &pshare[wid][0];

  // Q tile resident in registers (A layout)
  v16h aq[KC];
  {
    const _Float16* qrow = base + (size_t)(m0 + lm) * (3 * C) + head * DH + (hi << 3);
#pragma unroll
    for (int kc = 0; kc < KC; ++kc) {
      const v8h q0 = *(const v8h*)(qrow + kc * 32);
      const v8h q1 = *(const v8h*)(qrow + kc * 32 + 16);
#pragma unroll
      for (int h = 0; h < 8; ++h) { aq[kc][h] = q0[h]; aq[kc][h + 8] = q1[h]; }
    }
  }

  const v8f vzero = {};
  v8f oacc[DC];
  float rm[8], rl[8];
#pragma unroll
  for (int dc = 0; dc < DC; ++dc) oacc[dc] = vzero;
#pragma unroll
  for (int i = 0; i < 8; ++i) { rm[i] = -3.4e38f; rl[i] = 0.f; }

  for (int kt = 0; kt < Npts; kt += 32) {
    v8f sarr[2];
#pragma unroll
    for (int j = 0; j < 2; ++j) {
      v8f sa = vzero;
      const _Float16* krow =
          base + (size_t)(kt + j * 16 + lm) * (3 * C) + C + head * DH + (hi << 3);
#pragma unroll
      for (int kc = 0; kc < KC; ++kc) {
        const v8h k0 = *(const v8h*)(krow + kc * 32);
        const v8h k1 = *(const v8h*)(krow + kc * 32 + 16);
        v16h bk;
#pragma unroll
        for (int h = 0; h < 8; ++h) { bk[h] = k0[h]; bk[h + 8] = k1[h]; }
        sa = __builtin_amdgcn_wmma_f32_16x16x32_f16(false, aq[kc], false, bk,
                                                    (short)0, sa, false, false);
      }
      sarr[j] = sa;
    }

    // WAR: previous iteration's P reads must clear LDS before rewrite
    asm volatile("s_wait_dscnt 0" ::: "memory");

#pragma unroll
    for (int i = 0; i < 8; ++i) {
      const float v0 = sarr[0][i] * scale;
      const float v1 = sarr[1][i] * scale;
      float mx = fmaxf(v0, v1);
#pragma unroll
      for (int off = 1; off < 16; off <<= 1) mx = fmaxf(mx, __shfl_xor(mx, off, 32));
      const float mn = fmaxf(rm[i], mx);
      const float al = __expf(rm[i] - mn);
      rm[i] = mn;
      const float p0 = __expf(v0 - mn);
      const float p1 = __expf(v1 - mn);
      float ps = p0 + p1;
#pragma unroll
      for (int off = 1; off < 16; off <<= 1) ps += __shfl_xor(ps, off, 32);
      rl[i] = rl[i] * al + ps;
      const int mrow = i + (hi << 3);
      pb[mrow * 32 + lm]      = (_Float16)p0;
      pb[mrow * 32 + 16 + lm] = (_Float16)p1;
#pragma unroll
      for (int dc = 0; dc < DC; ++dc) oacc[dc][i] *= al;
    }

    // RAW: P must land in LDS before A-layout re-read
    asm volatile("s_wait_dscnt 0" ::: "memory");
    v16h pa;
    {
      const v8h p0 = *(const v8h*)(pb + lm * 32 + (hi << 3));
      const v8h p1 = *(const v8h*)(pb + lm * 32 + 16 + (hi << 3));
#pragma unroll
      for (int h = 0; h < 8; ++h) { pa[h] = p0[h]; pa[h + 8] = p1[h]; }
    }

#pragma unroll
    for (int dc = 0; dc < DC; ++dc) {
      const _Float16* vbase = base + 2 * C + head * DH + dc * 16 + lm;
      v16h bv;
#pragma unroll
      for (int h = 0; h < 16; ++h) {
        const int kk = ((h >> 3) << 4) + (hi << 3) + (h & 7);
        bv[h] = vbase[(size_t)(kt + kk) * (3 * C)];
      }
      oacc[dc] = __builtin_amdgcn_wmma_f32_16x16x32_f16(false, pa, false, bv,
                                                        (short)0, oacc[dc],
                                                        false, false);
    }
  }

  _Float16* ob = out + (size_t)b * Npts * C;
#pragma unroll
  for (int dc = 0; dc < DC; ++dc) {
    const int d = dc * 16 + lm;
#pragma unroll
    for (int i = 0; i < 8; ++i) {
      const int m = m0 + i + (hi << 3);
      ob[(size_t)m * C + head * DH + d] = (_Float16)(oacc[dc][i] / rl[i]);
    }
  }
}

// ---------------------------------------------------------------------------
// prob = softmax_n(H[b, :, o]);  pred[b,o,:] = sum_n prob * x[b,n,:]
// H: (B, N, Cout) f16; x,pred f32. Block per (b, o).
// ---------------------------------------------------------------------------
__global__ __launch_bounds__(256)
void softmax_pred_kernel(const _Float16* __restrict__ H, const float* __restrict__ X,
                         float* __restrict__ pred, int Npts, int Cout) {
  __shared__ float red[256];
  const int t = threadIdx.x;
  const int b = blockIdx.x / Cout;
  const int o = blockIdx.x % Cout;
  const _Float16* hb = H + (size_t)b * Npts * Cout + o;
  const float* xb = X + (size_t)b * Npts * 3;

  float mx = -3.4e38f;
  for (int n = t; n < Npts; n += 256) mx = fmaxf(mx, (float)hb[(size_t)n * Cout]);
  red[t] = mx; __syncthreads();
  for (int s = 128; s > 0; s >>= 1) { if (t < s) red[t] = fmaxf(red[t], red[t + s]); __syncthreads(); }
  mx = red[0]; __syncthreads();

  float zn = 0.f, sx = 0.f, sy = 0.f, sz = 0.f;
  for (int n = t; n < Npts; n += 256) {
    const float e = __expf((float)hb[(size_t)n * Cout] - mx);
    zn += e;
    sx += e * xb[n * 3 + 0];
    sy += e * xb[n * 3 + 1];
    sz += e * xb[n * 3 + 2];
  }
  float acc[4] = {zn, sx, sy, sz};
#pragma unroll
  for (int r = 0; r < 4; ++r) {
    red[t] = acc[r]; __syncthreads();
    for (int s = 128; s > 0; s >>= 1) { if (t < s) red[t] += red[t + s]; __syncthreads(); }
    acc[r] = red[0]; __syncthreads();
  }
  if (t == 0) {
    const float inv = 1.f / acc[0];
    float* pp = pred + ((size_t)b * Cout + o) * 3;
    pp[0] = acc[1] * inv; pp[1] = acc[2] * inv; pp[2] = acc[3] * inv;
  }
}

// ---------------------------------------------------------------------------
// Chamfer
// ---------------------------------------------------------------------------
__global__ __launch_bounds__(256)
void dist1_kernel(const float* __restrict__ pred, const float* __restrict__ gt,
                  float* __restrict__ d1, int NP, int MG) {
  __shared__ float red[256];
  const int t = threadIdx.x;
  const int b = blockIdx.x / NP;
  const int i = blockIdx.x % NP;
  const float* p = pred + ((size_t)b * NP + i) * 3;
  const float px = p[0], py = p[1], pz = p[2];
  const float* g = gt + (size_t)b * MG * 3;
  float best = 3.4e38f;
  for (int j = t; j < MG; j += 256) {
    const float dx = px - g[j * 3], dy = py - g[j * 3 + 1], dz = pz - g[j * 3 + 2];
    best = fminf(best, dx * dx + dy * dy + dz * dz);
  }
  red[t] = best; __syncthreads();
  for (int s = 128; s > 0; s >>= 1) { if (t < s) red[t] = fminf(red[t], red[t + s]); __syncthreads(); }
  if (t == 0) d1[blockIdx.x] = red[0];
}

__global__ __launch_bounds__(256)
void dist2_kernel(const float* __restrict__ pred, const float* __restrict__ gt,
                  float* __restrict__ d2, int NP, int MG) {
  const int idx = blockIdx.x * 256 + threadIdx.x;  // total B*MG, exact
  const int b = idx / MG;
  const float* g = gt + (size_t)idx * 3;
  const float gx = g[0], gy = g[1], gz = g[2];
  const float* p = pred + (size_t)b * NP * 3;
  float best = 3.4e38f;
  for (int i = 0; i < NP; ++i) {
    const float dx = gx - p[i * 3], dy = gy - p[i * 3 + 1], dz = gz - p[i * 3 + 2];
    best = fminf(best, dx * dx + dy * dy + dz * dz);
  }
  d2[idx] = best;
}

__global__ __launch_bounds__(256)
void cd_kernel(const float* __restrict__ d1, const float* __restrict__ d2,
               float* __restrict__ out, int NP, int MG) {
  __shared__ float red[256];
  const int t = threadIdx.x;
  const int b = blockIdx.x;
  float s1 = 0.f, q1 = 0.f, s2 = 0.f, q2 = 0.f;
  for (int i = t; i < NP; i += 256) { const float d = d1[b * NP + i]; s1 += sqrtf(d); q1 += d; }
  for (int j = t; j < MG; j += 256) { const float d = d2[b * MG + j]; s2 += sqrtf(d); q2 += d; }
  float acc[4] = {s1, q1, s2, q2};
#pragma unroll
  for (int r = 0; r < 4; ++r) {
    red[t] = acc[r]; __syncthreads();
    for (int s = 128; s > 0; s >>= 1) { if (t < s) red[t] += red[t + s]; __syncthreads(); }
    acc[r] = red[0]; __syncthreads();
  }
  if (t == 0) {
    out[b]     = 0.5f * (acc[0] / (float)NP + acc[2] / (float)MG);  // cd_l1
    out[8 + b] = acc[1] / (float)NP + acc[3] / (float)MG;           // cd_l2
  }
}

// ---------------------------------------------------------------------------
// Host side
// ---------------------------------------------------------------------------
struct LayerP {
  const float *proj_w, *proj_b, *n13_g, *n13_b, *in_w, *in_b, *out_w, *out_b,
              *n12_g, *n12_b, *l11_w, *l11_b, *l12_w, *l12_b;
};

static void launch_gemm(int act, const _Float16* A, const _Float16* W,
                        const float* bias, const _Float16* res, _Float16* C,
                        int M, int N, int K, hipStream_t s) {
  dim3 grid((unsigned)((M / 16) * (N / 64) / 8), BATCH);
  if (res) {
    if (act == 1) wmma_gemm_kernel<1, true><<<grid, 256, 0, s>>>(A, W, bias, res, C, M, N, K);
    else          wmma_gemm_kernel<0, true><<<grid, 256, 0, s>>>(A, W, bias, res, C, M, N, K);
  } else {
    if (act == 1) wmma_gemm_kernel<1, false><<<grid, 256, 0, s>>>(A, W, bias, nullptr, C, M, N, K);
    else          wmma_gemm_kernel<0, false><<<grid, 256, 0, s>>>(A, W, bias, nullptr, C, M, N, K);
  }
}

static void launch_cvt(const float* in, _Float16* out, int n, hipStream_t s) {
  cvt_f16_kernel<<<(n + 255) / 256, 256, 0, s>>>(in, out, n);
}

extern "C" void kernel_launch(void* const* d_in, const int* in_sizes, int n_in,
                              void* d_out, int out_size, void* d_ws, size_t ws_size,
                              hipStream_t stream) {
  (void)out_size; (void)ws_size;
  auto P = [&](int i) { return (const float*)d_in[i]; };

  const float *feat, *xpts, *gt;
  LayerP L[3];
  const bool insertion = (n_in >= 3) && (in_sizes[1] == BATCH * NPTS * 3);
  if (insertion) {  // top-level: feat, x, gt; per-layer in _make_layer order
    feat = P(0); xpts = P(1); gt = P(2);
    for (int l = 0; l < 3; ++l) {
      const int o = 3 + l * 14;
      L[l] = { P(o+0), P(o+1), P(o+2), P(o+3), P(o+4), P(o+5), P(o+6),
               P(o+7), P(o+8), P(o+9), P(o+10), P(o+11), P(o+12), P(o+13) };
    }
  } else {          // jax pytree alphabetical: feat, gt, params{...sorted...}, x
    feat = P(0); gt = P(1); xpts = P(n_in - 1);
    for (int l = 0; l < 3; ++l) {
      const int o = 2 + l * 14;
      L[l].in_b   = P(o+0);  L[l].in_w   = P(o+1);
      L[l].l11_b  = P(o+2);  L[l].l11_w  = P(o+3);
      L[l].l12_b  = P(o+4);  L[l].l12_w  = P(o+5);
      L[l].n12_b  = P(o+6);  L[l].n12_g  = P(o+7);
      L[l].n13_b  = P(o+8);  L[l].n13_g  = P(o+9);
      L[l].out_b  = P(o+10); L[l].out_w  = P(o+11);
      L[l].proj_b = P(o+12); L[l].proj_w = P(o+13);
    }
  }

  // Workspace layout (f16 activations + f16 weights + f32 dist arrays)
  const size_t S = (size_t)BATCH * NPTS * 1024;      // halves per max act slot
  _Float16* s0   = (_Float16*)d_ws;                  // input / attn_out / layer out
  _Float16* s1   = s0 + S;                           // x1 (post-LN), residual
  _Float16* qbuf = s1 + S;                           // qkv (3 slots) / FF hidden
  _Float16* wts  = qbuf + 3 * S;                     // converted weights
  float* d1 = (float*)(wts + 8388608);               // B*NOUT
  float* d2 = d1 + BATCH * NOUT;                     // B*NGT
  float* pred = (float*)d_out;
  float* cds  = pred + BATCH * NOUT * 3;

  // Convert all weights to f16 once per call; record per-layer pointers.
  _Float16 *proj_h[3], *in_h[3], *out_h[3], *l11_h[3], *l12_h[3];
  {
    _Float16* wp = wts;
    int cin = 256;
    for (int l = 0; l < 3; ++l) {
      const int cout = (l == 2) ? 1024 : 256;
      proj_h[l] = wp; launch_cvt(L[l].proj_w, wp, cout * cin, stream);      wp += cout * cin;
      in_h[l]   = wp; launch_cvt(L[l].in_w,   wp, 3 * cout * cout, stream); wp += 3 * cout * cout;
      out_h[l]  = wp; launch_cvt(L[l].out_w,  wp, cout * cout, stream);     wp += cout * cout;
      l11_h[l]  = wp; launch_cvt(L[l].l11_w,  wp, 1024 * cout, stream);     wp += 1024 * cout;
      l12_h[l]  = wp; launch_cvt(L[l].l12_w,  wp, cout * 1024, stream);     wp += cout * 1024;
      cin = cout;
    }
  }

  // feat (b,c,n) f32 -> s0 (b,n,c) f16
  transpose_cn_kernel<<<(BATCH * 256 * NPTS) / 256, 256, 0, stream>>>(feat, s0, BATCH, 256, NPTS);

  int cin = 256;
  for (int l = 0; l < 3; ++l) {
    const int cout = (l == 2) ? 1024 : 256;
    // 1. proj: s1 = s0 @ proj_w^T + proj_b
    launch_gemm(0, s0, proj_h[l], L[l].proj_b, nullptr, s1, NPTS, cout, cin, stream);
    // 2. LN(n13) in place -> x1
    layernorm_kernel<<<BATCH * NPTS, 256, 0, stream>>>(s1, L[l].n13_g, L[l].n13_b, cout);
    // 3. qkv = x1 @ in_w^T + in_b
    launch_gemm(0, s1, in_h[l], L[l].in_b, nullptr, qbuf, NPTS, 3 * cout, cout, stream);
    // 4. flash attention -> s0
    if (cout == 256)
      attention_kernel<64><<<dim3(NPTS / 128, 4, BATCH), 256, 0, stream>>>(qbuf, s0, NPTS, 0.125f);
    else
      attention_kernel<256><<<dim3(NPTS / 128, 4, BATCH), 256, 0, stream>>>(qbuf, s0, NPTS, 0.0625f);
    // 5. x1 = x1 + attn @ out_w^T + out_b  (in-place residual, elementwise-safe)
    launch_gemm(0, s0, out_h[l], L[l].out_b, s1, s1, NPTS, cout, cout, stream);
    // 6. LN(n12) in place
    layernorm_kernel<<<BATCH * NPTS, 256, 0, stream>>>(s1, L[l].n12_g, L[l].n12_b, cout);
    // 7. ff = GELU(x1 @ l11_w^T + l11_b) -> qbuf
    launch_gemm(1, s1, l11_h[l], L[l].l11_b, nullptr, qbuf, NPTS, 1024, cout, stream);
    // 8. out = x1 + ff @ l12_w^T + l12_b -> s0
    launch_gemm(0, qbuf, l12_h[l], L[l].l12_b, s1, s0, NPTS, cout, 1024, stream);
    cin = cout;
  }

  // softmax over points + matmul with x -> pred
  softmax_pred_kernel<<<BATCH * NOUT, 256, 0, stream>>>(s0, xpts, pred, NPTS, NOUT);
  // chamfer
  dist1_kernel<<<BATCH * NOUT, 256, 0, stream>>>(pred, gt, d1, NOUT, NGT);
  dist2_kernel<<<(BATCH * NGT) / 256, 256, 0, stream>>>(pred, gt, d2, NOUT, NGT);
  cd_kernel<<<BATCH, 256, 0, stream>>>(d1, d2, cds, NOUT, NGT);
}
